// TextRNN_36807869727477
// MI455X (gfx1250) — matrix operand order repaired and checked by reading, not compiled, not run on hardware
//
#include <hip/hip_runtime.h>

typedef __bf16 bf16_t;
typedef __attribute__((ext_vector_type(16))) __bf16 v16bf;
typedef __attribute__((ext_vector_type(8)))  __bf16 v8bf;
typedef __attribute__((ext_vector_type(8)))  float  v8f;

#define HID   1024
#define BATCH 64
#define SEQ   256
#define EMB   512
#define NCLS  32000

__device__ __forceinline__ float sigf(float x) { return 1.0f / (1.0f + __expf(-x)); }

// ---- WMMA fragment loaders (bf16, 16x16x32, wave32) -------------------------
// A (16x32, MxK): lanes 0-15 = rows, khalf = lane>>4.
//   elements 0..7  -> K = k0 + khalf*16 + e
//   elements 8..15 -> K = k0 + khalf*16 + 8 + (e-8)
__device__ __forceinline__ v16bf load_a_frag(const bf16_t* row, int k0, int lane) {
  int khalf = lane >> 4;
  union { v16bf v; v8bf h[2]; } u;
  u.h[0] = *(const v8bf*)(row + k0 + khalf * 16);
  u.h[1] = *(const v8bf*)(row + k0 + khalf * 16 + 8);
  return u.v;
}
// B (32x16, KxN) sourced from B^T storage (row n contiguous in K):
//   element e -> K = k0 + khalf*16 + e  => one contiguous 32B load
__device__ __forceinline__ v16bf load_b_frag(const bf16_t* rowT, int k0, int lane) {
  int khalf = lane >> 4;
  return *(const v16bf*)(rowT + k0 + khalf * 16);
}

// ---- prep kernels -----------------------------------------------------------
__global__ __launch_bounds__(256) void embed_kernel(const int* __restrict__ X,
                                                    const float* __restrict__ C,
                                                    bf16_t* __restrict__ xs) {
  size_t idx = (size_t)blockIdx.x * blockDim.x + threadIdx.x;      // [t][b][e]
  if (idx >= (size_t)SEQ * BATCH * EMB) return;
  int e = idx & (EMB - 1);
  int b = (idx >> 9) & (BATCH - 1);
  int t = (int)(idx >> 15);
  int tok = X[b * SEQ + t];
  xs[idx] = (bf16_t)C[(size_t)tok * EMB + e];
}

// Pack 4 x-side + 4 h-side weight matrices into B^T bf16 [4096][Dx+1024]
// gate order (rows n): [0,1024)=g, [1024,2048)=i, [2048,3072)=f, [3072,4096)=o
__global__ __launch_bounds__(256) void pack_lstm_weights(
    const float* __restrict__ Wgx, const float* __restrict__ Wix,
    const float* __restrict__ Wfx, const float* __restrict__ Wox,
    const float* __restrict__ Wgh, const float* __restrict__ Wih,
    const float* __restrict__ Wfh, const float* __restrict__ Woh,
    const float* __restrict__ bg,  const float* __restrict__ bi,
    const float* __restrict__ bf_, const float* __restrict__ bo,
    int Dx, bf16_t* __restrict__ Wt, float* __restrict__ bias) {
  const int K = Dx + HID;
  size_t idx = (size_t)blockIdx.x * blockDim.x + threadIdx.x;
  if (idx >= (size_t)4 * HID * K) return;
  int k = (int)(idx % K);
  int n = (int)(idx / K);
  int gate = n >> 10;
  int col  = n & (HID - 1);
  const float* Wx[4] = {Wgx, Wix, Wfx, Wox};
  const float* Wh[4] = {Wgh, Wih, Wfh, Woh};
  float v = (k < Dx) ? Wx[gate][(size_t)k * HID + col]
                     : Wh[gate][(size_t)(k - Dx) * HID + col];
  Wt[idx] = (bf16_t)v;
  if (k == 0) {
    const float* bs[4] = {bg, bi, bf_, bo};
    bias[n] = bs[gate][col];
  }
}

__global__ __launch_bounds__(256) void pack_w3(const float* __restrict__ W3,
                                               bf16_t* __restrict__ W3t) {
  size_t idx = (size_t)blockIdx.x * blockDim.x + threadIdx.x;      // [n][k]
  if (idx >= (size_t)NCLS * HID) return;
  int k = idx & (HID - 1);
  int n = (int)(idx >> 10);
  W3t[idx] = (bf16_t)W3[(size_t)k * NCLS + n];
}

__global__ __launch_bounds__(256) void init_state(const float* __restrict__ h0,
                                                  const float* __restrict__ c0,
                                                  bf16_t* __restrict__ h,
                                                  float* __restrict__ c) {
  int idx = blockIdx.x * blockDim.x + threadIdx.x;
  if (idx >= BATCH * HID) return;
  h[idx] = (bf16_t)h0[idx];
  c[idx] = c0[idx];
}

// ---- fused LSTM step --------------------------------------------------------
// Block owns one M tile (16 batch rows) and 8 N tiles (one per wave).
// A slab [16][K] = [x_t | h_prev] staged once in LDS (branch-free K loop, 8x
// less A traffic, ds broadcast reads). Each wave keeps 4 WMMA accumulators
// (g,i,f,o) sharing one A fragment, with a 1-deep software pipeline: next
// k-step's A (LDS) + 4 B (global) fragments are in flight while the current 4
// WMMAs execute, so waits become partial instead of s_wait_loadcnt 0.
__global__ __launch_bounds__(256) void lstm_step_kernel(
    const bf16_t* __restrict__ x_t, int Dx,
    const bf16_t* __restrict__ Wt, const float* __restrict__ bias,
    const bf16_t* __restrict__ hprev,
    bf16_t* __restrict__ hnew, float* __restrict__ cstate,
    bf16_t* __restrict__ hs_out) {
  extern __shared__ char smem[];
  bf16_t* As = (bf16_t*)smem;                 // [16][K] row-major
  const int K = Dx + HID;

  const int mt     = blockIdx.x >> 3;         // 0..3 (M tile)
  const int ntbase = (blockIdx.x & 7) * 8;    // 8 waves -> nt = ntbase + waveid
  const int lane = threadIdx.x & 31;
  const int nt   = ntbase + (threadIdx.x >> 5);
  const int l15  = lane & 15;

  // ---- cooperative A-slab stage: rows mt*16..+15 of [x_t | h_prev] ----------
  {
    const int cx = (16 * Dx) >> 3;            // 8-elem (16B) chunks of x part
    for (int c = threadIdx.x; c < cx; c += 256) {
      int row = c / (Dx >> 3);
      int col = (c - row * (Dx >> 3)) << 3;
      *(v8bf*)(As + row * K + col) =
          *(const v8bf*)(x_t + (size_t)(mt * 16 + row) * Dx + col);
    }
    const int ch = (16 * HID) >> 3;           // chunks of h part
    for (int c = threadIdx.x; c < ch; c += 256) {
      int row = c >> 7;                       // HID/8 = 128 chunks per row
      int col = (c & 127) << 3;
      *(v8bf*)(As + row * K + Dx + col) =
          *(const v8bf*)(hprev + (size_t)(mt * 16 + row) * HID + col);
    }
  }
  __syncthreads();

  const bf16_t* arow = As + l15 * K;
  const bf16_t* brow[4];
#pragma unroll
  for (int g = 0; g < 4; ++g)
    brow[g] = Wt + (size_t)(g * HID + nt * 16 + l15) * K;

  v8f zero = {0.f, 0.f, 0.f, 0.f, 0.f, 0.f, 0.f, 0.f};
  v8f acc[4] = {zero, zero, zero, zero};

  // prologue of the 1-deep pipeline
  v16bf a_cur = load_a_frag(arow, 0, lane);
  v16bf b_cur[4];
#pragma unroll
  for (int g = 0; g < 4; ++g) b_cur[g] = load_b_frag(brow[g], 0, lane);

#pragma unroll 2
  for (int k0 = 0; k0 < K; k0 += 32) {
    // prefetch next k-step (last iter over-reads: LDS OOB -> 0, global lands
    // in adjacent ws carve; both unused)
    v16bf a_nxt = load_a_frag(arow, k0 + 32, lane);
    v16bf b_nxt[4];
#pragma unroll
    for (int g = 0; g < 4; ++g) b_nxt[g] = load_b_frag(brow[g], k0 + 32, lane);

#pragma unroll
    for (int g = 0; g < 4; ++g)
      acc[g] = __builtin_amdgcn_wmma_f32_16x16x32_bf16(
          false, a_cur, false, b_cur[g], (short)0, acc[g], false, false);

    a_cur = a_nxt;
#pragma unroll
    for (int g = 0; g < 4; ++g) b_cur[g] = b_nxt[g];
  }

  const int khalf = lane >> 4;                // D layout: VGPR r -> M = khalf*8+r
  const int n = nt * 16 + l15;
#pragma unroll
  for (int r = 0; r < 8; ++r) {
    int m = mt * 16 + khalf * 8 + r;
    float g = tanhf(acc[0][r] + bias[n]);
    float i = sigf (acc[1][r] + bias[HID + n]);
    float f = sigf (acc[2][r] + bias[2 * HID + n]);
    float o = sigf (acc[3][r] + bias[3 * HID + n]);
    size_t idx = (size_t)m * HID + n;
    float c = cstate[idx] * f + g * i;
    cstate[idx] = c;
    float h = tanhf(c) * o;
    hnew[idx] = (bf16_t)h;
    if (hs_out) hs_out[idx] = (bf16_t)h;
  }
}

// ---- classifier: out[64,32000] = hT @ W3 + b2 ------------------------------
// Block owns one M tile; 8 waves cover 8 consecutive N tiles. hT M-tile staged
// in LDS (32 KB); same 1-deep software pipeline as the step kernel.
__global__ __launch_bounds__(256) void classifier_kernel(
    const bf16_t* __restrict__ hT, const bf16_t* __restrict__ W3t,
    const float* __restrict__ b2, float* __restrict__ out) {
  __shared__ bf16_t As[16 * HID + 32];        // +32: benign prefetch over-read

  const int mt     = blockIdx.x / 250;        // 0..3
  const int ntbase = (blockIdx.x % 250) * 8;  // nt = ntbase + waveid (0..1999)
  const int lane = threadIdx.x & 31;
  const int nt   = ntbase + (threadIdx.x >> 5);
  const int l15  = lane & 15;

  for (int c = threadIdx.x; c < (16 * HID) >> 3; c += 256) {
    int row = c >> 7;
    int col = (c & 127) << 3;
    *(v8bf*)(As + row * HID + col) =
        *(const v8bf*)(hT + (size_t)(mt * 16 + row) * HID + col);
  }
  __syncthreads();

  const bf16_t* arow = As + l15 * HID;
  const bf16_t* brow = W3t + (size_t)(nt * 16 + l15) * HID;

  v8f acc = {0.f, 0.f, 0.f, 0.f, 0.f, 0.f, 0.f, 0.f};
  v16bf a_cur = load_a_frag(arow, 0, lane);
  v16bf b_cur = load_b_frag(brow, 0, lane);
#pragma unroll 2
  for (int k0 = 0; k0 < HID; k0 += 32) {
    v16bf a_nxt = load_a_frag(arow, k0 + 32, lane);
    v16bf b_nxt = load_b_frag(brow, k0 + 32, lane);
    acc = __builtin_amdgcn_wmma_f32_16x16x32_bf16(
        false, a_cur, false, b_cur, (short)0, acc, false, false);
    a_cur = a_nxt;
    b_cur = b_nxt;
  }
  const int khalf = lane >> 4;
  const int n = nt * 16 + l15;
#pragma unroll
  for (int r = 0; r < 8; ++r) {
    int m = mt * 16 + khalf * 8 + r;
    out[(size_t)m * NCLS + n] = acc[r] + b2[n];
  }
}

// ---- host orchestration -----------------------------------------------------
extern "C" void kernel_launch(void* const* d_in, const int* in_sizes, int n_in,
                              void* d_out, int out_size, void* d_ws, size_t ws_size,
                              hipStream_t stream) {
  const int*   X    = (const int*)d_in[0];
  const float* C    = (const float*)d_in[1];
  const float* W1   = (const float*)d_in[2];
  const float* Wxi  = (const float*)d_in[3];
  const float* Wxf  = (const float*)d_in[4];
  const float* Wxo  = (const float*)d_in[5];
  const float* W2   = (const float*)d_in[6];
  const float* Whi  = (const float*)d_in[7];
  const float* Whf  = (const float*)d_in[8];
  const float* Who  = (const float*)d_in[9];
  const float* b1   = (const float*)d_in[10];
  const float* b_i  = (const float*)d_in[11];
  const float* b_f  = (const float*)d_in[12];
  const float* b_o  = (const float*)d_in[13];
  const float* W1y  = (const float*)d_in[14];
  const float* Wxiy = (const float*)d_in[15];
  const float* Wxfy = (const float*)d_in[16];
  const float* Wxoy = (const float*)d_in[17];
  const float* W2y  = (const float*)d_in[18];
  const float* Whiy = (const float*)d_in[19];
  const float* Whfy = (const float*)d_in[20];
  const float* Whoy = (const float*)d_in[21];
  const float* b1y  = (const float*)d_in[22];
  const float* b_iy = (const float*)d_in[23];
  const float* b_fy = (const float*)d_in[24];
  const float* b_oy = (const float*)d_in[25];
  const float* W3   = (const float*)d_in[26];
  const float* b2   = (const float*)d_in[27];
  const float* h0   = (const float*)d_in[28];
  const float* c0   = (const float*)d_in[29];
  const float* h0y  = (const float*)d_in[30];
  const float* c0y  = (const float*)d_in[31];
  float* out = (float*)d_out;

  // workspace carve-out (256B aligned)
  char* wsp = (char*)d_ws;
  auto carve = [&](size_t bytes) -> void* {
    void* p = (void*)wsp;
    wsp += (bytes + 255) & ~(size_t)255;
    return p;
  };
  bf16_t* xs1   = (bf16_t*)carve((size_t)SEQ * BATCH * EMB * 2);       // 16 MB
  bf16_t* hs    = (bf16_t*)carve((size_t)SEQ * BATCH * HID * 2);       // 32 MB
  bf16_t* Wt1   = (bf16_t*)carve((size_t)4 * HID * (EMB + HID) * 2);   // 12 MB
  bf16_t* Wt2   = (bf16_t*)carve((size_t)4 * HID * (HID + HID) * 2);   // 16 MB
  bf16_t* W3t   = (bf16_t*)carve((size_t)NCLS * HID * 2);              // 64 MB
  float*  bias1 = (float*)carve((size_t)4 * HID * 4);
  float*  bias2 = (float*)carve((size_t)4 * HID * 4);
  bf16_t* h1    = (bf16_t*)carve((size_t)2 * BATCH * HID * 2);         // double buffer
  bf16_t* h2    = (bf16_t*)carve((size_t)2 * BATCH * HID * 2);
  float*  c1    = (float*)carve((size_t)BATCH * HID * 4);
  float*  c2    = (float*)carve((size_t)BATCH * HID * 4);

  const int TPB = 256;
  // one-time prep: embedding gather + fp32->bf16 weight repack (B^T layout)
  {
    size_t n = (size_t)SEQ * BATCH * EMB;
    embed_kernel<<<(unsigned)((n + TPB - 1) / TPB), TPB, 0, stream>>>(X, C, xs1);
  }
  {
    size_t n = (size_t)4 * HID * (EMB + HID);
    pack_lstm_weights<<<(unsigned)((n + TPB - 1) / TPB), TPB, 0, stream>>>(
        W1, Wxi, Wxf, Wxo, W2, Whi, Whf, Who, b1, b_i, b_f, b_o, EMB, Wt1, bias1);
  }
  {
    size_t n = (size_t)4 * HID * (HID + HID);
    pack_lstm_weights<<<(unsigned)((n + TPB - 1) / TPB), TPB, 0, stream>>>(
        W1y, Wxiy, Wxfy, Wxoy, W2y, Whiy, Whfy, Whoy, b1y, b_iy, b_fy, b_oy,
        HID, Wt2, bias2);
  }
  {
    size_t n = (size_t)NCLS * HID;
    pack_w3<<<(unsigned)((n + TPB - 1) / TPB), TPB, 0, stream>>>(W3, W3t);
  }
  init_state<<<(BATCH * HID + TPB - 1) / TPB, TPB, 0, stream>>>(h0, c0, h1, c1);
  init_state<<<(BATCH * HID + TPB - 1) / TPB, TPB, 0, stream>>>(h0y, c0y, h2, c2);

  const size_t HB = (size_t)BATCH * HID;              // elements per h buffer
  const size_t lds1 = (size_t)16 * (EMB + HID) * 2 + 64;  // A slab + prefetch pad
  const size_t lds2 = (size_t)16 * (HID + HID) * 2 + 64;

  // layer 1: 256 dependent steps, weights (12 MB bf16) resident in L2
  for (int t = 0; t < SEQ; ++t) {
    lstm_step_kernel<<<32, 256, lds1, stream>>>(
        xs1 + (size_t)t * BATCH * EMB, EMB, Wt1, bias1,
        h1 + (size_t)(t & 1) * HB, h1 + (size_t)((t + 1) & 1) * HB, c1,
        hs + (size_t)t * BATCH * HID);
  }
  // layer 2: consumes layer-1 h stream (bf16) directly
  for (int t = 0; t < SEQ; ++t) {
    lstm_step_kernel<<<32, 256, lds2, stream>>>(
        hs + (size_t)t * BATCH * HID, HID, Wt2, bias2,
        h2 + (size_t)(t & 1) * HB, h2 + (size_t)((t + 1) & 1) * HB, c2,
        (bf16_t*)nullptr);
  }
  // final h is in h2 buffer parity 0 (t=255 wrote (255+1)&1 == 0)
  classifier_kernel<<<(4 * (NCLS / 16)) / 8, 256, 0, stream>>>(h2, W3t, b2, out);
}
